// PointNetFeaturePropagation_54168127537473
// MI455X (gfx1250) — matrix operand order repaired and checked by reading, not compiled, not run on hardware
//
#include <hip/hip_runtime.h>
#include <hip/hip_bf16.h>

// Problem constants (from reference setup_inputs)
#define B_    8
#define N_    8192
#define S_    2048
#define DUNK  128
#define DKN   256
#define C0IN  384
#define C0OUT 256
#define C1OUT 128
#define NTOT  (B_ * N_)          // 65536 points total
#define BN_EPS   1e-5f
#define DIST_EPS 1e-8f

// GEMM block tile: 128 rows x 64 cols, 8 waves in 4(M) x 2(N) grid,
// each wave owns a 2x2 grid of 16x16 WMMA accumulators.
#define BM 128
#define BN 64
#define LDS_STRIDE 20            // dwords per LDS tile row (80B, 16B aligned)

typedef __attribute__((ext_vector_type(16))) __bf16 v16bf;
typedef __attribute__((ext_vector_type(8)))  float  v8f;

// ---------- bf16 helpers (round-to-nearest-even) ----------
__device__ __forceinline__ unsigned short f2bf(float x) {
  unsigned int u = __float_as_uint(x);
  unsigned int r = u + 0x7FFFu + ((u >> 16) & 1u);
  return (unsigned short)(r >> 16);
}
__device__ __forceinline__ float bf2f(unsigned int h16) {
  return __uint_as_float((h16 & 0xFFFFu) << 16);
}
__device__ __forceinline__ unsigned int pack2(float lo, float hi) {
  return (unsigned int)f2bf(lo) | ((unsigned int)f2bf(hi) << 16);
}

// ---------- WMMA fragment loads (ISA 7.12.2, 16-bit layouts) ----------
// A 16x32: lanes 0-15 M=lane, K 0..7 (v0-3) & 16..23 (v4-7); lanes 16-31 same M, K+8.
//   -> per lane: dwords [lg*4 .. lg*4+3] and [8+lg*4 .. 8+lg*4+3]  (two b128 loads)
// B 32x16: lane%16 = N col; same K order as A (K 0..7 & 16..23 for lanes 0-15, K+8 for 16-31).
//   -> per lane: dwords [lg*4 .. lg*4+3] and [8+lg*4 .. 8+lg*4+3]  (two b128 loads)
__device__ __forceinline__ v16bf load_fragA(const unsigned int* A, int row, int lg) {
  union { uint4 q[2]; v16bf v; } f;
  const unsigned int* p = A + row * LDS_STRIDE;
  f.q[0] = *(const uint4*)(p + lg * 4);
  f.q[1] = *(const uint4*)(p + 8 + lg * 4);
  return f.v;
}
__device__ __forceinline__ v16bf load_fragB(const unsigned int* Bt, int col, int lg) {
  union { uint4 q[2]; v16bf v; } f;
  const unsigned int* p = Bt + col * LDS_STRIDE;
  f.q[0] = *(const uint4*)(p + lg * 4);
  f.q[1] = *(const uint4*)(p + 8 + lg * 4);
  return f.v;
}

// ---------- kernel 1: convert weights to bf16, zero stat accumulators ----------
__global__ void init_convert(const float* __restrict__ W0, const float* __restrict__ W1,
                             unsigned short* __restrict__ W0b, unsigned short* __restrict__ W1b,
                             float* __restrict__ stats, int n0, int n1, int ns) {
  int total = n0 + n1 + ns;
  int stride = gridDim.x * blockDim.x;
  for (int i = blockIdx.x * blockDim.x + threadIdx.x; i < total; i += stride) {
    if (i < n0)            W0b[i] = f2bf(W0[i]);
    else if (i < n0 + n1)  W1b[i - n0] = f2bf(W1[i - n0]);
    else                   stats[i - n0 - n1] = 0.0f;
  }
}

// ---------- kernel 2: 3-NN + inverse-distance interp + concat -> bf16 xcat ----------
// grid (N_/256, B_), block 256.  known_xyz of one batch staged in LDS (24KB).
__global__ __launch_bounds__(256)
void knn_interp_concat(const float* __restrict__ uxyz, const float* __restrict__ kxyz,
                       const float* __restrict__ upts, const float* __restrict__ kpts,
                       unsigned int* __restrict__ xcat_u) {
  __shared__ float sk[S_ * 3];
  const int b   = blockIdx.y;
  const int tid = threadIdx.x;
  const float* kb = kxyz + (size_t)b * S_ * 3;
  for (int i = tid; i < S_ * 3; i += blockDim.x) sk[i] = kb[i];
  __syncthreads();

  const int n = blockIdx.x * blockDim.x + tid;
  const size_t row = (size_t)b * N_ + n;
  const float* up = uxyz + row * 3;
  const float ux = up[0], uy = up[1], uz = up[2];

  float d0 = 3.4e38f, d1 = 3.4e38f, d2 = 3.4e38f;
  int   i0 = 0, i1 = 0, i2 = 0;
  #pragma unroll 2
  for (int s = 0; s < S_; ++s) {
    float dx = ux - sk[3 * s + 0];
    float dy = uy - sk[3 * s + 1];
    float dz = uz - sk[3 * s + 2];
    float d  = fmaf(dx, dx, fmaf(dy, dy, dz * dz));
    if (d < d2) {
      if (d < d1) {
        if (d < d0) { d2 = d1; i2 = i1; d1 = d0; i1 = i0; d0 = d; i0 = s; }
        else        { d2 = d1; i2 = i1; d1 = d;  i1 = s; }
      } else        { d2 = d;  i2 = s; }
    }
  }
  float w0 = 1.0f / (d0 + DIST_EPS);
  float w1 = 1.0f / (d1 + DIST_EPS);
  float w2 = 1.0f / (d2 + DIST_EPS);
  const float wn = 1.0f / (w0 + w1 + w2);
  w0 *= wn; w1 *= wn; w2 *= wn;

  unsigned int* xo = xcat_u + row * (C0IN / 2);
  uint2* xo2 = (uint2*)xo;
  // first 128 channels: copy unknown_points (fp32 -> bf16), float4 at a time
  const float4* uf4 = (const float4*)(upts + row * DUNK);
  #pragma unroll 4
  for (int c = 0; c < DUNK / 4; ++c) {
    float4 v = uf4[c];
    xo2[c] = make_uint2(pack2(v.x, v.y), pack2(v.z, v.w));
  }
  // next 256 channels: weighted sum of 3 neighbor feature rows, float4 at a time
  const float4* q0 = (const float4*)(kpts + ((size_t)b * S_ + i0) * DKN);
  const float4* q1 = (const float4*)(kpts + ((size_t)b * S_ + i1) * DKN);
  const float4* q2 = (const float4*)(kpts + ((size_t)b * S_ + i2) * DKN);
  #pragma unroll 4
  for (int c = 0; c < DKN / 4; ++c) {
    float4 a = q0[c], bq = q1[c], dq = q2[c];
    float x = fmaf(w0, a.x, fmaf(w1, bq.x, w2 * dq.x));
    float y = fmaf(w0, a.y, fmaf(w1, bq.y, w2 * dq.y));
    float z = fmaf(w0, a.z, fmaf(w1, bq.z, w2 * dq.z));
    float w = fmaf(w0, a.w, fmaf(w1, bq.w, w2 * dq.w));
    xo2[DUNK / 4 + c] = make_uint2(pack2(x, y), pack2(z, w));
  }
}

// ---------- kernel 3: GEMM0  y0 = xcat(bf16) @ W0^T + b0 -> bf16 ----------
// M=65536, N=256, K=384.  grid (M/128, N/64), block 256.
__global__ __launch_bounds__(256)
void gemm0_wmma(const unsigned int* __restrict__ Xu,   // [M][K/2] bf16 pairs
                const unsigned int* __restrict__ Wu,   // [256][K/2]
                const float* __restrict__ bias,
                unsigned short* __restrict__ Yb) {     // [M][256] bf16
  __shared__ unsigned int ldsA[2][BM * LDS_STRIDE];
  __shared__ unsigned int ldsB[2][BN * LDS_STRIDE];
  const int tid = threadIdx.x;
  const int Mbase = blockIdx.x * BM;
  const int Nbase = blockIdx.y * BN;
  const int lane = tid & 31, wave = tid >> 5;
  const int wm = wave >> 1, wn = wave & 1;        // 4 x 2 wave grid
  const int lg = lane >> 4, lm = lane & 15;
  const int KU = C0IN / 2;                         // 192 dwords per row
  const int NC = C0IN / 32;                        // 12 K-chunks

  auto stage = [&](int buf, int kc) {
    unsigned int* A  = ldsA[buf];
    unsigned int* Bt = ldsB[buf];
    #pragma unroll
    for (int t = 0; t < 2; ++t) {                  // A tile: 128x16 dwords as b128
      int u = tid + t * 256, r = u >> 2, c4 = (u & 3) * 4;
      uint4 v = *(const uint4*)&Xu[(size_t)(Mbase + r) * KU + kc * 16 + c4];
      *(uint4*)&A[r * LDS_STRIDE + c4] = v;
    }
    {                                              // B tile: 64x16 dwords as b128
      int r = tid >> 2, c4 = (tid & 3) * 4;
      uint4 v = *(const uint4*)&Wu[(size_t)(Nbase + r) * KU + kc * 16 + c4];
      *(uint4*)&Bt[r * LDS_STRIDE + c4] = v;
    }
  };

  v8f acc[2][2];
  #pragma unroll
  for (int sm = 0; sm < 2; ++sm)
    #pragma unroll
    for (int sn = 0; sn < 2; ++sn)
      acc[sm][sn] = (v8f){0.f, 0.f, 0.f, 0.f, 0.f, 0.f, 0.f, 0.f};

  stage(0, 0);
  for (int kc = 0; kc < NC; ++kc) {
    __syncthreads();
    if (kc + 1 < NC) stage((kc + 1) & 1, kc + 1);
    if (kc + 2 < NC)                               // gfx1250 global_prefetch_b8
      __builtin_prefetch(&Xu[(size_t)(Mbase + (tid >> 2)) * KU + (kc + 2) * 16], 0, 1);
    const unsigned int* A  = ldsA[kc & 1];
    const unsigned int* Bt = ldsB[kc & 1];
    v16bf af[2], bf_[2];
    #pragma unroll
    for (int sm = 0; sm < 2; ++sm) af[sm]  = load_fragA(A, wm * 32 + sm * 16 + lm, lg);
    #pragma unroll
    for (int sn = 0; sn < 2; ++sn) bf_[sn] = load_fragB(Bt, wn * 32 + sn * 16 + lm, lg);
    #pragma unroll
    for (int sm = 0; sm < 2; ++sm)
      #pragma unroll
      for (int sn = 0; sn < 2; ++sn)
        acc[sm][sn] = __builtin_amdgcn_wmma_f32_16x16x32_bf16(
            false, af[sm], false, bf_[sn], (short)0, acc[sm][sn], false, false);
  }

  #pragma unroll
  for (int sn = 0; sn < 2; ++sn) {
    const int col = Nbase + wn * 32 + sn * 16 + lm;
    const float bv = bias[col];
    #pragma unroll
    for (int sm = 0; sm < 2; ++sm) {
      const int rowBase = Mbase + wm * 32 + sm * 16 + lg * 8;  // lanes 16-31 -> M+8
      #pragma unroll
      for (int r = 0; r < 8; ++r)
        Yb[(size_t)(rowBase + r) * C0OUT + col] = f2bf(acc[sm][sn][r] + bv);
    }
  }
}

// ---------- kernel 4a: per-channel sum/sumsq over bf16 tensor (channel-strided) ----------
// each thread owns one bf16 channel-pair, accumulates over strided rows in registers.
__global__ void bn_stats_bf16(const unsigned int* __restrict__ Yu, int rows, int Chalf,
                              float* __restrict__ gsum, float* __restrict__ gsq) {
  const int gtid  = blockIdx.x * blockDim.x + threadIdx.x;
  const int total = gridDim.x * blockDim.x;
  const int cp = gtid % Chalf;
  const int r0 = gtid / Chalf;
  const int rs = total / Chalf;
  float slo = 0.f, qlo = 0.f, shi = 0.f, qhi = 0.f;
  for (int r = r0; r < rows; r += rs) {
    unsigned int raw = Yu[(size_t)r * Chalf + cp];
    float lo = bf2f(raw), hi = bf2f(raw >> 16);
    slo += lo; qlo += lo * lo;
    shi += hi; qhi += hi * hi;
  }
  atomicAdd(&gsum[2 * cp],     slo); atomicAdd(&gsq[2 * cp],     qlo);
  atomicAdd(&gsum[2 * cp + 1], shi); atomicAdd(&gsq[2 * cp + 1], qhi);
}

// ---------- kernel 4b: per-channel sum/sumsq over fp32 tensor (channel-strided) ----------
__global__ void bn_stats_f32(const float* __restrict__ Y, int rows, int C,
                             float* __restrict__ gsum, float* __restrict__ gsq) {
  const int gtid  = blockIdx.x * blockDim.x + threadIdx.x;
  const int total = gridDim.x * blockDim.x;
  const int c  = gtid % C;
  const int r0 = gtid / C;
  const int rs = total / C;
  float s = 0.f, q = 0.f;
  for (int r = r0; r < rows; r += rs) {
    float v = Y[(size_t)r * C + c];
    s += v; q += v * v;
  }
  atomicAdd(&gsum[c], s); atomicAdd(&gsq[c], q);
}

// ---------- kernel 5: fold BN into per-channel scale/shift ----------
__global__ void bn_finalize(const float* __restrict__ sum, const float* __restrict__ sq,
                            const float* __restrict__ g, const float* __restrict__ beta,
                            float* __restrict__ scale, float* __restrict__ shift,
                            int C, float invCnt) {
  int c = blockIdx.x * blockDim.x + threadIdx.x;
  if (c < C) {
    float m  = sum[c] * invCnt;
    float v  = sq[c] * invCnt - m * m;
    float rs = rsqrtf(v + BN_EPS);
    float s  = g[c] * rs;
    scale[c] = s;
    shift[c] = beta[c] - m * s;
  }
}

// ---------- kernel 6: GEMM1 with fused BN0+ReLU on A ----------
// A = relu(bn0(y0)) bf16 on the fly; M=65536, N=128, K=256. grid (M/128, N/64).
__global__ __launch_bounds__(256)
void gemm1_wmma(const unsigned short* __restrict__ Yb,  // y0 bf16 [M][256]
                const unsigned int* __restrict__ Wu,    // W1 bf16 [128][128 dwords]
                const float* __restrict__ bias,
                const float* __restrict__ scale0,
                const float* __restrict__ shift0,
                float* __restrict__ Z) {                // y1 fp32 [M][128]
  __shared__ unsigned int ldsA[2][BM * LDS_STRIDE];
  __shared__ unsigned int ldsB[2][BN * LDS_STRIDE];
  __shared__ float ssc[C0OUT], ssh[C0OUT];
  const unsigned int* Yu = (const unsigned int*)Yb;
  const int tid = threadIdx.x;
  const int Mbase = blockIdx.x * BM;
  const int Nbase = blockIdx.y * BN;
  const int lane = tid & 31, wave = tid >> 5;
  const int wm = wave >> 1, wn = wave & 1;
  const int lg = lane >> 4, lm = lane & 15;
  const int KU = C0OUT / 2;                        // 128 dwords per y0 row
  const int NC = C0OUT / 32;                       // 8 K-chunks

  for (int c = tid; c < C0OUT; c += 256) { ssc[c] = scale0[c]; ssh[c] = shift0[c]; }
  __syncthreads();

  auto stage = [&](int buf, int kc) {
    unsigned int* A  = ldsA[buf];
    unsigned int* Bt = ldsB[buf];
    #pragma unroll
    for (int t = 0; t < 2; ++t) {                  // A tile with fused BN0 + ReLU
      int u = tid + t * 256, r = u >> 2, c4 = (u & 3) * 4;
      uint4 v = *(const uint4*)&Yu[(size_t)(Mbase + r) * KU + kc * 16 + c4];
      const int ch = kc * 32 + c4 * 2;
      unsigned int w[4];
      unsigned int raw[4] = {v.x, v.y, v.z, v.w};
      #pragma unroll
      for (int j = 0; j < 4; ++j) {
        float lo = fmaxf(0.f, fmaf(bf2f(raw[j]),       ssc[ch + 2 * j],     ssh[ch + 2 * j]));
        float hi = fmaxf(0.f, fmaf(bf2f(raw[j] >> 16), ssc[ch + 2 * j + 1], ssh[ch + 2 * j + 1]));
        w[j] = pack2(lo, hi);
      }
      *(uint4*)&A[r * LDS_STRIDE + c4] = make_uint4(w[0], w[1], w[2], w[3]);
    }
    {                                              // B tile
      int r = tid >> 2, c4 = (tid & 3) * 4;
      uint4 v = *(const uint4*)&Wu[(size_t)(Nbase + r) * KU + kc * 16 + c4];
      *(uint4*)&Bt[r * LDS_STRIDE + c4] = v;
    }
  };

  v8f acc[2][2];
  #pragma unroll
  for (int sm = 0; sm < 2; ++sm)
    #pragma unroll
    for (int sn = 0; sn < 2; ++sn)
      acc[sm][sn] = (v8f){0.f, 0.f, 0.f, 0.f, 0.f, 0.f, 0.f, 0.f};

  stage(0, 0);
  for (int kc = 0; kc < NC; ++kc) {
    __syncthreads();
    if (kc + 1 < NC) stage((kc + 1) & 1, kc + 1);
    if (kc + 2 < NC)
      __builtin_prefetch(&Yu[(size_t)(Mbase + (tid >> 2)) * KU + (kc + 2) * 16], 0, 1);
    const unsigned int* A  = ldsA[kc & 1];
    const unsigned int* Bt = ldsB[kc & 1];
    v16bf af[2], bf_[2];
    #pragma unroll
    for (int sm = 0; sm < 2; ++sm) af[sm]  = load_fragA(A, wm * 32 + sm * 16 + lm, lg);
    #pragma unroll
    for (int sn = 0; sn < 2; ++sn) bf_[sn] = load_fragB(Bt, wn * 32 + sn * 16 + lm, lg);
    #pragma unroll
    for (int sm = 0; sm < 2; ++sm)
      #pragma unroll
      for (int sn = 0; sn < 2; ++sn)
        acc[sm][sn] = __builtin_amdgcn_wmma_f32_16x16x32_bf16(
            false, af[sm], false, bf_[sn], (short)0, acc[sm][sn], false, false);
  }

  #pragma unroll
  for (int sn = 0; sn < 2; ++sn) {
    const int col = Nbase + wn * 32 + sn * 16 + lm;
    const float bv = bias[col];
    #pragma unroll
    for (int sm = 0; sm < 2; ++sm) {
      const int rowBase = Mbase + wm * 32 + sm * 16 + lg * 8;
      #pragma unroll
      for (int r = 0; r < 8; ++r)
        Z[(size_t)(rowBase + r) * C1OUT + col] = acc[sm][sn][r] + bv;
    }
  }
}

// ---------- kernel 7: apply BN1 + ReLU -> final output (float4) ----------
__global__ void bn_apply(const float4* __restrict__ Y, const float* __restrict__ scale,
                         const float* __restrict__ shift, float4* __restrict__ out,
                         long n4, int C) {
  const long stride = (long)gridDim.x * blockDim.x;
  for (long i = (long)blockIdx.x * blockDim.x + threadIdx.x; i < n4; i += stride) {
    float4 v = Y[i];
    int c = (int)((i * 4) % C);
    out[i] = make_float4(fmaxf(0.f, fmaf(v.x, scale[c],     shift[c])),
                         fmaxf(0.f, fmaf(v.y, scale[c + 1], shift[c + 1])),
                         fmaxf(0.f, fmaf(v.z, scale[c + 2], shift[c + 2])),
                         fmaxf(0.f, fmaf(v.w, scale[c + 3], shift[c + 3])));
  }
}

// ---------- workspace layout ----------
#define OFF_W0B   0                                    // 256*384 bf16
#define OFF_W1B   (OFF_W0B + 256 * 384 * 2)            // 128*256 bf16
#define OFF_XCAT  (OFF_W1B + 128 * 256 * 2)            // 65536*384 bf16
#define OFF_Y0    (OFF_XCAT + (size_t)NTOT * C0IN * 2) // 65536*256 bf16
#define OFF_Y1    (OFF_Y0 + (size_t)NTOT * C0OUT * 2)  // 65536*128 fp32
#define OFF_STATS (OFF_Y1 + (size_t)NTOT * C1OUT * 4)  // 1536 floats
#define WS_NEEDED (OFF_STATS + 1536 * 4)

extern "C" void kernel_launch(void* const* d_in, const int* in_sizes, int n_in,
                              void* d_out, int out_size, void* d_ws, size_t ws_size,
                              hipStream_t stream) {
  const float* uxyz  = (const float*)d_in[0];
  const float* kxyz  = (const float*)d_in[1];
  const float* upts  = (const float*)d_in[2];
  const float* kpts  = (const float*)d_in[3];
  const float* W0    = (const float*)d_in[4];
  const float* b0    = (const float*)d_in[5];
  const float* g0    = (const float*)d_in[6];
  const float* beta0 = (const float*)d_in[7];
  const float* W1    = (const float*)d_in[8];
  const float* b1    = (const float*)d_in[9];
  const float* g1    = (const float*)d_in[10];
  const float* beta1 = (const float*)d_in[11];
  (void)in_sizes; (void)n_in; (void)out_size;
  if (ws_size < WS_NEEDED) return;

  char* ws = (char*)d_ws;
  unsigned short* W0b  = (unsigned short*)(ws + OFF_W0B);
  unsigned short* W1b  = (unsigned short*)(ws + OFF_W1B);
  unsigned int*   xcat = (unsigned int*)(ws + OFF_XCAT);
  unsigned short* y0   = (unsigned short*)(ws + OFF_Y0);
  float*          y1   = (float*)(ws + OFF_Y1);
  float* stats  = (float*)(ws + OFF_STATS);
  float* sum0   = stats;        float* sq0    = stats + 256;
  float* scale0 = stats + 512;  float* shift0 = stats + 768;
  float* sum1   = stats + 1024; float* sq1    = stats + 1152;
  float* scale1 = stats + 1280; float* shift1 = stats + 1408;

  init_convert<<<dim3(512), dim3(256), 0, stream>>>(
      W0, W1, W0b, W1b, stats, 256 * 384, 128 * 256, 1536);

  knn_interp_concat<<<dim3(N_ / 256, B_), dim3(256), 0, stream>>>(
      uxyz, kxyz, upts, kpts, xcat);

  gemm0_wmma<<<dim3(NTOT / BM, C0OUT / BN), dim3(256), 0, stream>>>(
      xcat, (const unsigned int*)W0b, b0, y0);

  bn_stats_bf16<<<dim3(256), dim3(256), 0, stream>>>(
      (const unsigned int*)y0, NTOT, C0OUT / 2, sum0, sq0);

  bn_finalize<<<dim3(1), dim3(256), 0, stream>>>(
      sum0, sq0, g0, beta0, scale0, shift0, C0OUT, 1.0f / (float)NTOT);

  gemm1_wmma<<<dim3(NTOT / BM, C1OUT / BN), dim3(256), 0, stream>>>(
      y0, (const unsigned int*)W1b, b1, scale0, shift0, y1);

  bn_stats_f32<<<dim3(256), dim3(256), 0, stream>>>(
      y1, NTOT, C1OUT, sum1, sq1);

  bn_finalize<<<dim3(1), dim3(256), 0, stream>>>(
      sum1, sq1, g1, beta1, scale1, shift1, C1OUT, 1.0f / (float)NTOT);

  bn_apply<<<dim3(2048), dim3(256), 0, stream>>>(
      (const float4*)y1, scale1, shift1, (float4*)d_out,
      (long)NTOT * C1OUT / 4, C1OUT);
}
